// RDMModel_87385404605013
// MI455X (gfx1250) — matrix operand order repaired and checked by reading, not compiled, not run on hardware
//
#include <hip/hip_runtime.h>
#include <hip/hip_bf16.h>

// Batched Pearson correlation for [B=32, N=512, D=2048] f32.
// Pass 1: per-row mean + 1/sqrt(centered sum of squares) into workspace.
// Pass 2: batched SYRK via split-bf16 (hi+lo) WMMA (3x v_wmma_f32_16x16x32_bf16
//         per tile per K-chunk -> ~f32 accuracy at bf16 matrix throughput),
//         fused normalize + clip epilogue.

typedef __attribute__((ext_vector_type(16))) __bf16 bf16x16;
typedef __attribute__((ext_vector_type(8)))  __bf16 bf16x8;
typedef __attribute__((ext_vector_type(8)))  float  f32x8;
typedef __attribute__((ext_vector_type(4)))  float  f32x4;

#define BATCH 32
#define NVAR  512
#define DDIM  2048
#define KB    32          // K elements staged per LDS stage (one WMMA K-chunk)
#define PITCH 40          // KB + 8 pad (bf16 elems) = 80B: 16B-aligned rows,
                          // 20-bank row stride -> conflict-free b128 frag loads
#define TILE  128         // output tile per workgroup (128x128)

union frag_u { bf16x16 v; bf16x8 h[2]; };

// ---------------------------------------------------------------- pass 1 ----
__global__ __launch_bounds__(256)
void row_stats_kernel(const float* __restrict__ x,
                      float* __restrict__ mean,
                      float* __restrict__ rstd) {
  const int wave = threadIdx.x >> 5;
  const int lane = threadIdx.x & 31;
  const int row  = blockIdx.x * 8 + wave;            // 0 .. B*N-1
  const float* p = x + (size_t)row * DDIM;

  float s1 = 0.0f, s2 = 0.0f;
  for (int k = lane * 4; k < DDIM; k += 32 * 4) {    // b128 loads per lane
    f32x4 v = *(const f32x4*)(p + k);
#pragma unroll
    for (int e = 0; e < 4; ++e) { s1 += v[e]; s2 += v[e] * v[e]; }
  }
#pragma unroll
  for (int m = 16; m >= 1; m >>= 1) {                // wave32 butterfly reduce
    s1 += __shfl_xor(s1, m, 32);
    s2 += __shfl_xor(s2, m, 32);
  }
  if (lane == 0) {
    float mu = s1 * (1.0f / (float)DDIM);
    float ss = s2 - (float)DDIM * mu * mu;           // = diag(cov) in exact math
    mean[row] = mu;
    rstd[row] = rsqrtf(ss);
  }
}

// ---------------------------------------------------------------- pass 2 ----
__global__ __launch_bounds__(256)
void corr_syrk_kernel(const float* __restrict__ x,
                      const float* __restrict__ mean,
                      const float* __restrict__ rstd,
                      float* __restrict__ out) {
  __shared__ __bf16 sAhi[TILE * PITCH];
  __shared__ __bf16 sAlo[TILE * PITCH];
  __shared__ __bf16 sBhi[TILE * PITCH];
  __shared__ __bf16 sBlo[TILE * PITCH];

  const int b    = blockIdx.y;
  const int i0   = (blockIdx.x >> 2) * TILE;         // row block
  const int j0   = (blockIdx.x & 3)  * TILE;         // col block
  const int tid  = threadIdx.x;
  const int wave = tid >> 5;
  const int lane = tid & 31;
  const int wm   = wave >> 1;                        // 0..3: 32-row strip
  const int wn   = wave & 1;                         // 0..1: 64-col strip
  const int half = lane >> 4;                        // lane group 0/1
  const int l16  = lane & 15;

  const float* xb = x    + (size_t)b * NVAR * DDIM;
  const float* mb = mean + b * NVAR;
  const float* rb = rstd + b * NVAR;

  f32x8 acc[2][4];
#pragma unroll
  for (int it = 0; it < 2; ++it)
#pragma unroll
    for (int jt = 0; jt < 4; ++jt)
      acc[it][jt] = (f32x8){0.f, 0.f, 0.f, 0.f, 0.f, 0.f, 0.f, 0.f};

  for (int k0 = 0; k0 < DDIM; k0 += KB) {
    __syncthreads();
    // ---- stage: center f32 -> split into bf16 hi/lo slabs in LDS ----------
#pragma unroll
    for (int s = 0; s < 2; ++s) {
      const int base = (s == 0) ? i0 : j0;
      __bf16* hi = (s == 0) ? sAhi : sBhi;
      __bf16* lo = (s == 0) ? sAlo : sBlo;
#pragma unroll
      for (int it = 0; it < 4; ++it) {               // 1024 float4 / 256 thr
        const int idx = it * 256 + tid;
        const int row = idx >> 3;                    // 8 float4 per row
        const int kq  = idx & 7;
        f32x4 v = *(const f32x4*)(xb + (size_t)(base + row) * DDIM + k0 + kq * 4);
        const float mu = mb[base + row];
        __bf16* ph = hi + row * PITCH + kq * 4;
        __bf16* pl = lo + row * PITCH + kq * 4;
#pragma unroll
        for (int e = 0; e < 4; ++e) {
          float xc = v[e] - mu;
          __bf16 h = (__bf16)xc;                     // hi part (RNE)
          ph[e] = h;
          pl[e] = (__bf16)(xc - (float)h);           // residual lo part
        }
      }
    }
    __syncthreads();

    // ---- A fragments: ISA 16-bit 16x32 A layout ---------------------------
    // lanes 0-15: K 0-7 & 16-23 ; lanes 16-31: K 8-15 & 24-31
    bf16x16 aHi[2], aLo[2];
#pragma unroll
    for (int it = 0; it < 2; ++it) {
      const int r = wm * 32 + it * 16 + l16;
      frag_u u;
      const __bf16* pa = &sAhi[r * PITCH + half * 8];
      u.h[0] = *(const bf16x8*)(pa);
      u.h[1] = *(const bf16x8*)(pa + 16);
      aHi[it] = u.v;
      const __bf16* pl = &sAlo[r * PITCH + half * 8];
      u.h[0] = *(const bf16x8*)(pl);
      u.h[1] = *(const bf16x8*)(pl + 16);
      aLo[it] = u.v;
    }

    // ---- B fragments + 3 WMMAs per output tile ----------------------------
    // lanes 0-15: col N=l16, K 0-7 & 16-23 ; lanes 16-31: col N=l16, K 8-15 & 24-31
#pragma unroll
    for (int jt = 0; jt < 4; ++jt) {
      const int c = wn * 64 + jt * 16 + l16;
      frag_u u;
      const __bf16* pb = &sBhi[c * PITCH + half * 8];
      u.h[0] = *(const bf16x8*)(pb);
      u.h[1] = *(const bf16x8*)(pb + 16);
      bf16x16 bHi = u.v;
      const __bf16* pl = &sBlo[c * PITCH + half * 8];
      u.h[0] = *(const bf16x8*)(pl);
      u.h[1] = *(const bf16x8*)(pl + 16);
      bf16x16 bLo = u.v;
#pragma unroll
      for (int it = 0; it < 2; ++it) {
        acc[it][jt] = __builtin_amdgcn_wmma_f32_16x16x32_bf16(
            false, aHi[it], false, bHi, (short)0, acc[it][jt], false, false);
        acc[it][jt] = __builtin_amdgcn_wmma_f32_16x16x32_bf16(
            false, aHi[it], false, bLo, (short)0, acc[it][jt], false, false);
        acc[it][jt] = __builtin_amdgcn_wmma_f32_16x16x32_bf16(
            false, aLo[it], false, bHi, (short)0, acc[it][jt], false, false);
      }
    }
  }

  // ---- epilogue: corr = cov * rstd_i * rstd_j, clip to [-1, 1] ------------
  // C/D layout: VGPR r holds M = r (lanes 0-15) / M = r+8 (lanes 16-31), N = l16
#pragma unroll
  for (int it = 0; it < 2; ++it) {
#pragma unroll
    for (int jt = 0; jt < 4; ++jt) {
      const int gj = j0 + wn * 64 + jt * 16 + l16;
      const float rj = rb[gj];
#pragma unroll
      for (int r = 0; r < 8; ++r) {
        const int gi = i0 + wm * 32 + it * 16 + r + half * 8;
        float v = acc[it][jt][r] * rb[gi] * rj;
        v = fminf(1.0f, fmaxf(-1.0f, v));
        out[((size_t)b * NVAR + gi) * NVAR + gj] = v;
      }
    }
  }
}

// ---------------------------------------------------------------- launch ----
extern "C" void kernel_launch(void* const* d_in, const int* in_sizes, int n_in,
                              void* d_out, int out_size, void* d_ws, size_t ws_size,
                              hipStream_t stream) {
  const float* x = (const float*)d_in[0];
  float* out  = (float*)d_out;
  float* mean = (float*)d_ws;                  // B*N floats
  float* rstd = mean + BATCH * NVAR;           // B*N floats (128 KB total)

  row_stats_kernel<<<dim3((BATCH * NVAR) / 8), dim3(256), 0, stream>>>(x, mean, rstd);
  corr_syrk_kernel<<<dim3(16, BATCH), dim3(256), 0, stream>>>(x, mean, rstd, out);
}
